// SmallBitNet_70617852281040
// MI455X (gfx1250) — matrix operand order repaired and checked by reading, not compiled, not run on hardware
//
#include <hip/hip_runtime.h>
#include <hip/hip_bf16.h>

typedef _Float16 v16h __attribute__((ext_vector_type(16)));
typedef _Float16 v8h  __attribute__((ext_vector_type(8)));
typedef _Float16 v4h  __attribute__((ext_vector_type(4)));
typedef float    v8f  __attribute__((ext_vector_type(8)));
typedef float    v4f  __attribute__((ext_vector_type(4)));

#define DIM   1024
#define TOKS  65536
#define GS    16
#define NGRP  (DIM / GS)          // 64 groups per output row
#define NLIN  12
#define LN_EPS 1e-5f

// ---------------------------------------------------------------------------
// Dequantize int4 codes (stored as int) * group scale -> f16 weights, natural
// [layer][o][k] row-major layout (exactly what the B-fragment loads want).
// One thread per group of 16 input channels.
// ---------------------------------------------------------------------------
__global__ __launch_bounds__(256)
void dequant4b_kernel(const int* __restrict__ qw,
                      const float* __restrict__ scales,
                      _Float16* __restrict__ W)
{
    const int idx = blockIdx.x * blockDim.x + threadIdx.x;   // [0, 12*1024*64)
    const float s = scales[idx];
    const size_t base = (size_t)idx * GS;
    v16h w;
#pragma unroll
    for (int j = 0; j < GS; ++j)
        w[j] = (_Float16)((float)qw[base + j] * s);
    *(v16h*)(W + base) = w;
}

// ---------------------------------------------------------------------------
// f32 -> f16 conversion of the input activations (4 elements / thread)
// ---------------------------------------------------------------------------
__global__ __launch_bounds__(256)
void cvt_f32_f16_kernel(const float* __restrict__ X, _Float16* __restrict__ Y)
{
    const size_t i = (size_t)(blockIdx.x * blockDim.x + threadIdx.x) * 4;
    v4f v = *(const v4f*)(X + i);
    v4h h;
#pragma unroll
    for (int j = 0; j < 4; ++j) h[j] = (_Float16)v[j];
    *(v4h*)(Y + i) = h;
}

// ---------------------------------------------------------------------------
// WMMA GEMM: O[t][o] = act( A[t][:] . W[o][:] + bias[o] (+ R[t][o]) )
//   A: [TOKS][1024] f16 row-major, W: [1024][1024] f16 row-major (this layer)
//   Block: 256 threads = 8 waves -> 128x256 tile; wave tile 64x64 (4x4 wmma)
//   => 16 global_load_b128 feed 16 v_wmma per K-chunk (1.0 load/wmma).
// CDNA5 fragment layouts (CDNA5 ISA):
//   A 16x32 f16: lane L (r=L&15,hi=L>>4): halves 0..7  = A[r][8*hi+0..7]
//                                         halves 8..15 = A[r][16+8*hi+0..7]
//   B 32x16 f16: lane L: column N=r, halves 0..7 -> K=8*hi+j, 8..15 -> K=16+8*hi+(j-8)
//   D 16x16 f32: lane L: N=r, vgpr p -> M = p + 8*hi
// ---------------------------------------------------------------------------
template<bool RELU, bool RESID>
__global__ __launch_bounds__(256)
void gemm_wmma_kernel(const _Float16* __restrict__ A,
                      const _Float16* __restrict__ W,
                      const float*    __restrict__ bias,
                      const _Float16* __restrict__ R,
                      _Float16*       __restrict__ O)
{
    const int lane = threadIdx.x & 31;
    const int wave = threadIdx.x >> 5;
    const int wm   = wave >> 2;           // 0..1
    const int wn   = wave & 3;            // 0..3
    const int r    = lane & 15;
    const int hi   = lane >> 4;

    const int tm0 = blockIdx.y * 128 + wm * 64;   // token base (4 x 16-row subtiles)
    const int on0 = blockIdx.x * 256 + wn * 64;   // output base (4 x 16-col subtiles)

    v8f acc[4][4] = {};

    const _Float16* arow[4];
#pragma unroll
    for (int i = 0; i < 4; ++i)
        arow[i] = A + (size_t)(tm0 + 16 * i + r) * DIM;
    const _Float16* wrow[4];
#pragma unroll
    for (int j = 0; j < 4; ++j)
        wrow[j] = W + (size_t)(on0 + 16 * j + r) * DIM;

    for (int kc = 0; kc < DIM; kc += 32) {
        union { v16h v; v8h h[2]; } af[4], bf[4];
#pragma unroll
        for (int i = 0; i < 4; ++i) {
            af[i].h[0] = *(const v8h*)(arow[i] + kc +      8 * hi);
            af[i].h[1] = *(const v8h*)(arow[i] + kc + 16 + 8 * hi);
        }
#pragma unroll
        for (int j = 0; j < 4; ++j) {
            bf[j].h[0] = *(const v8h*)(wrow[j] + kc +      8 * hi);
            bf[j].h[1] = *(const v8h*)(wrow[j] + kc + 16 + 8 * hi);
        }

#pragma unroll
        for (int i = 0; i < 4; ++i)
#pragma unroll
            for (int j = 0; j < 4; ++j)
                acc[i][j] = __builtin_amdgcn_wmma_f32_16x16x32_f16(
                    /*neg_a=*/false, af[i].v, /*neg_b=*/false, bf[j].v,
                    /*c_mod=*/(short)0, acc[i][j],
                    /*reuse_a=*/false, /*reuse_b=*/false);
    }

    // epilogue: bias (+relu) (+residual), store f16
    float bj[4];
#pragma unroll
    for (int j = 0; j < 4; ++j) bj[j] = bias[on0 + 16 * j + r];

#pragma unroll
    for (int i = 0; i < 4; ++i) {
        const int trow = tm0 + 16 * i + 8 * hi;
#pragma unroll
        for (int j = 0; j < 4; ++j) {
            const int o = on0 + 16 * j + r;
#pragma unroll
            for (int p = 0; p < 8; ++p) {
                const size_t idx = (size_t)(trow + p) * DIM + o;
                float v = acc[i][j][p] + bj[j];
                if (RELU)  v = v > 0.f ? v : 0.f;
                if (RESID) v += (float)R[idx];
                O[idx] = (_Float16)v;
            }
        }
    }
}

// ---------------------------------------------------------------------------
// Row LayerNorm, in place over f16 buffer; optional f32 final output.
// One block (256 threads, 8 waves) per token row of 1024.
// ---------------------------------------------------------------------------
__global__ __launch_bounds__(256)
void lnorm_kernel(_Float16* __restrict__ H,
                  const float* __restrict__ lw,
                  const float* __restrict__ lb,
                  float* __restrict__ finalOut)
{
    const size_t row = blockIdx.x;
    _Float16* hr = H + row * DIM;

    float v[4], s1 = 0.f, s2 = 0.f;
#pragma unroll
    for (int j = 0; j < 4; ++j) {
        v[j] = (float)hr[threadIdx.x + 256 * j];
        s1 += v[j];
        s2 += v[j] * v[j];
    }
#pragma unroll
    for (int m = 16; m >= 1; m >>= 1) {
        s1 += __shfl_xor(s1, m, 32);
        s2 += __shfl_xor(s2, m, 32);
    }
    __shared__ float p1[8], p2[8];
    if ((threadIdx.x & 31) == 0) {
        p1[threadIdx.x >> 5] = s1;
        p2[threadIdx.x >> 5] = s2;
    }
    __syncthreads();
    float t1 = 0.f, t2 = 0.f;
#pragma unroll
    for (int k = 0; k < 8; ++k) { t1 += p1[k]; t2 += p2[k]; }
    const float mu  = t1 * (1.f / DIM);
    const float var = t2 * (1.f / DIM) - mu * mu;
    const float rs  = rsqrtf(var + LN_EPS);

#pragma unroll
    for (int j = 0; j < 4; ++j) {
        const int c = threadIdx.x + 256 * j;
        const float y = (v[j] - mu) * rs * lw[c] + lb[c];
        hr[c] = (_Float16)y;
        if (finalOut) finalOut[row * DIM + c] = y;
    }
}

// ---------------------------------------------------------------------------
// Host launcher. Workspace layout (all f16):
//   W  : 12*1024*1024            dequantized weights (24 MB, L2 resident)
//   X,Y,Z : 65536*1024 each      activation ping-pong buffers (128 MB each)
// Per block b: X --l0/relu--> Y --l1/relu--> Z --l2+resid(X)--> X --LN(inplace)
// ---------------------------------------------------------------------------
extern "C" void kernel_launch(void* const* d_in, const int* in_sizes, int n_in,
                              void* d_out, int out_size, void* d_ws, size_t ws_size,
                              hipStream_t stream)
{
    const float* x      = (const float*)d_in[0];
    const int*   qw     = (const int*)  d_in[1];
    const float* scales = (const float*)d_in[2];
    const float* bias   = (const float*)d_in[3];
    const float* ln_w   = (const float*)d_in[4];
    const float* ln_b   = (const float*)d_in[5];
    float* out = (float*)d_out;

    _Float16* W = (_Float16*)d_ws;
    _Float16* X = W + (size_t)NLIN * DIM * DIM;
    _Float16* Y = X + (size_t)TOKS * DIM;
    _Float16* Z = Y + (size_t)TOKS * DIM;

    // 1) dequantize weights once (12*1024*64 groups)
    dequant4b_kernel<<<NLIN * DIM * NGRP / 256, 256, 0, stream>>>(qw, scales, W);
    // 2) f32 -> f16 input conversion
    cvt_f32_f16_kernel<<<TOKS * DIM / (256 * 4), 256, 0, stream>>>(x, X);

    const dim3 gg(DIM / 256, TOKS / 128);   // (4, 512)
    for (int blk = 0; blk < 4; ++blk) {
        const _Float16* W0 = W + (size_t)(3 * blk + 0) * DIM * DIM;
        const _Float16* W1 = W + (size_t)(3 * blk + 1) * DIM * DIM;
        const _Float16* W2 = W + (size_t)(3 * blk + 2) * DIM * DIM;

        gemm_wmma_kernel<true,  false><<<gg, 256, 0, stream>>>(
            X, W0, bias + (size_t)(3 * blk + 0) * DIM, nullptr, Y);
        gemm_wmma_kernel<true,  false><<<gg, 256, 0, stream>>>(
            Y, W1, bias + (size_t)(3 * blk + 1) * DIM, nullptr, Z);
        gemm_wmma_kernel<false, true ><<<gg, 256, 0, stream>>>(
            Z, W2, bias + (size_t)(3 * blk + 2) * DIM, X, X);

        lnorm_kernel<<<TOKS, 256, 0, stream>>>(
            X, ln_w + (size_t)blk * DIM, ln_b + (size_t)blk * DIM,
            (blk == 3) ? out : nullptr);
    }
}
